// HyperbolicAE_14688788152450
// MI455X (gfx1250) — matrix operand order repaired and checked by reading, not compiled, not run on hardware
//
#include <hip/hip_runtime.h>
#include <hip/hip_bf16.h>

typedef __attribute__((ext_vector_type(16))) _Float16 v16h;
typedef __attribute__((ext_vector_type(8)))  float    v8f;

#define NNODE   64
#define DDIM    64
#define NPAIRS  8192      // B*N
#define NEDGES  524288    // B*N*N

__device__ inline v8f wmma_f16(v16h a, v16h b, v8f c) {
  // D = A(16x32 f16) * B(32x16 f16) + C(16x16 f32)
  return __builtin_amdgcn_wmma_f32_16x16x32_f16(false, a, false, b, (short)0, c,
                                                false, false);
}

__device__ inline float siluf(float x) { return x / (1.f + __expf(-x)); }

__device__ inline float softplusf(float x) {
  if (x > 20.f) return x;
  if (x < -20.f) return __expf(x);
  return log1pf(__expf(x));
}

// Build one B-fragment (32x16, f16) from a row-major 64x64 f32 matrix.
// ISA layout: lanes 0-15 hold K=k0+{0..7,16..23} of column n, lanes 16-31 K=k0+{8..15,24..31}.
__device__ inline v16h load_bfrag(const float* __restrict__ Wp, int lane, int t, int c) {
  int n  = t * 16 + (lane & 15);
  int k0 = c * 32 + ((lane >> 4) << 3);
  v16h b;
#pragma unroll
  for (int v = 0; v < 16; ++v)
    b[v] = (_Float16)Wp[(k0 + (v < 8 ? v : v + 8)) * DDIM + n];
  return b;
}

__global__ void init_kernel(float* __restrict__ accum) {
  accum[0] = 0.f;
  accum[1] = 0.f;
}

// Apre = h @ W1[0:64,:], Bpre = h @ W1[64:128,:]   (each 8192x64, f32 out)
// One wave per (16-row group, half). WMMA f16 with f32 accumulate.
__global__ __launch_bounds__(32) void precompute_kernel(
    const float* __restrict__ h, const float* __restrict__ W1,
    float* __restrict__ outA, float* __restrict__ outB) {
  int wid  = blockIdx.x;           // 0..1023
  int sel  = wid & 1;              // 0: W1 top half, 1: bottom half
  int r0   = (wid >> 1) * 16;      // row group
  int lane = threadIdx.x;
  int hs   = lane >> 4;            // lane half
  int base = hs * 8;               // A-fragment K base per ISA layout
  int mrow = r0 + (lane & 15);

  const float* Wp   = W1 + sel * DDIM * DDIM;
  float*       outp = sel ? outB : outA;

  // Resident W1-half B-fragments: 4 N-tiles x 2 K-chunks
  v16h bf[4][2];
#pragma unroll
  for (int t = 0; t < 4; ++t)
#pragma unroll
    for (int c = 0; c < 2; ++c)
      bf[t][c] = load_bfrag(Wp, lane, t, c);

  // A-fragments from h row (f32 -> f16), K = 0..63 split in 2 chunks
  const float* hp = h + mrow * DDIM;
  v16h af[2];
#pragma unroll
  for (int c = 0; c < 2; ++c) {
    v16h aa;
#pragma unroll
    for (int j = 0; j < 16; ++j) {
      int k = base + 32 * c + (j < 8 ? j : j + 8);
      aa[j] = (_Float16)hp[k];
    }
    af[c] = aa;
  }

  v8f zero = {0.f, 0.f, 0.f, 0.f, 0.f, 0.f, 0.f, 0.f};
  v8f acc[4] = {zero, zero, zero, zero};
#pragma unroll
  for (int t = 0; t < 4; ++t) {
    acc[t] = wmma_f16(af[0], bf[t][0], acc[t]);
    acc[t] = wmma_f16(af[1], bf[t][1], acc[t]);
  }

  // Store per ISA C/D layout: element (m,n): m = v + 8*hs, n = 16t + lane&15
#pragma unroll
  for (int t = 0; t < 4; ++t) {
    int n = t * 16 + (lane & 15);
#pragma unroll
    for (int v = 0; v < 8; ++v)
      outp[(r0 + v + 8 * hs) * DDIM + n] = acc[t][v];
  }
}

// One wave handles 8 (b,i) rows x 4 col-tiles of 16 edges each.
// z1 = silu(Apre[row] + Bpre[col] + b1) built directly in A-fragment layout,
// layer 2 via WMMA, layer 3 + softplus + squared error via VALU/shfl.
__global__ __launch_bounds__(32) void edge_kernel(
    const float* __restrict__ Apre, const float* __restrict__ Bpre,
    const float* __restrict__ b1,  const float* __restrict__ W2,
    const float* __restrict__ b2,  const float* __restrict__ W3,
    const float* __restrict__ b3,  const float* __restrict__ tgt,
    const float* __restrict__ emask, float* __restrict__ accum) {
  int lane = threadIdx.x;
  int hs   = lane >> 4;
  int base = hs * 8;
  int lcol = lane & 15;

  // Resident W2 B-fragments (layer 2 weights)
  v16h bf[4][2];
#pragma unroll
  for (int t = 0; t < 4; ++t)
#pragma unroll
    for (int c = 0; c < 2; ++c)
      bf[t][c] = load_bfrag(W2, lane, t, c);

  // Per-lane layer-3 constants: n = 16t + lcol
  float w3v[4], b2v[4];
#pragma unroll
  for (int t = 0; t < 4; ++t) {
    int n = t * 16 + lcol;
    w3v[t] = W3[n];
    b2v[t] = b2[n];
  }
  float b3s = b3[0];

  float sumsq = 0.f;
  v8f zero = {0.f, 0.f, 0.f, 0.f, 0.f, 0.f, 0.f, 0.f};

  int p0 = blockIdx.x * 8;
  for (int pp = 0; pp < 8; ++pp) {
    int p    = p0 + pp;       // row node index = b*64 + i
    int bblk = p >> 6;
    int i    = p & 63;

    // Hoist Apre[row] + b1 at this lane's K positions (4 runs of 8)
    float aih[4][8];
    const float* Ap = Apre + p * DDIM;
#pragma unroll
    for (int r = 0; r < 4; ++r) {
      int ks = base + 16 * r;
#pragma unroll
      for (int u = 0; u < 8; ++u)
        aih[r][u] = Ap[ks + u] + b1[ks + u];
    }

    for (int j0 = 0; j0 < NNODE; j0 += 16) {
      // z1 for 16 edges (cols j0..j0+15): lane's edge-row is lcol
      const float* Bp = Bpre + (bblk * NNODE + j0 + lcol) * DDIM;
      float run[4][8];
#pragma unroll
      for (int r = 0; r < 4; ++r) {
        int ks = base + 16 * r;
#pragma unroll
        for (int u = 0; u < 8; ++u) {
          float x = aih[r][u] + Bp[ks + u];
          run[r][u] = siluf(x);
        }
      }
      v16h af[2];
#pragma unroll
      for (int c = 0; c < 2; ++c) {
        v16h aa;
#pragma unroll
        for (int j = 0; j < 16; ++j)
          aa[j] = (_Float16)(j < 8 ? run[2 * c][j] : run[2 * c + 1][j - 8]);
        af[c] = aa;
      }

      v8f acc[4] = {zero, zero, zero, zero};
#pragma unroll
      for (int t = 0; t < 4; ++t) {
        acc[t] = wmma_f16(af[0], bf[t][0], acc[t]);
        acc[t] = wmma_f16(af[1], bf[t][1], acc[t]);
      }

      // Layer 3: q[v] = sum_t silu(z2 + b2[n]) * W3[n] ; rows m = v + 8*hs
      float q[8];
#pragma unroll
      for (int v = 0; v < 8; ++v) {
        float s = 0.f;
#pragma unroll
        for (int t = 0; t < 4; ++t)
          s += siluf(acc[t][v] + b2v[t]) * w3v[t];
        q[v] = s;
      }
      // Sum over the 16 lanes in each half (n sweep)
#pragma unroll
      for (int v = 0; v < 8; ++v) {
        q[v] += __shfl_xor(q[v], 1, 32);
        q[v] += __shfl_xor(q[v], 2, 32);
        q[v] += __shfl_xor(q[v], 4, 32);
        q[v] += __shfl_xor(q[v], 8, 32);
      }
      // Lane (lcol = v) of each half finalizes edge m = v + 8*hs
      float myDot = q[0];
#pragma unroll
      for (int v = 1; v < 8; ++v)
        if (lcol == v) myDot = q[v];
      if (lcol < 8) {
        int m = lcol + 8 * hs;
        int e = bblk * (NNODE * NNODE) + i * NNODE + j0 + m;
        float mk = emask[e];
        float d  = softplusf(myDot + b3s) * mk - tgt[e] * mk;
        sumsq += d * d;
      }
    }
  }

  // Wave reduction + one atomic
  sumsq += __shfl_xor(sumsq, 1, 32);
  sumsq += __shfl_xor(sumsq, 2, 32);
  sumsq += __shfl_xor(sumsq, 4, 32);
  sumsq += __shfl_xor(sumsq, 8, 32);
  sumsq += __shfl_xor(sumsq, 16, 32);
  if (lane == 0) atomicAdd(accum, sumsq);
}

__global__ void rec_kernel(const float* __restrict__ xhat,
                           const int* __restrict__ cat,
                           const float* __restrict__ nmask,
                           float* __restrict__ accum) {
  int idx = blockIdx.x * blockDim.x + threadIdx.x;
  float ce = 0.f;
  if (idx < NPAIRS) {
    const float* x = xhat + idx * 6;
    float mx = x[0];
#pragma unroll
    for (int k = 1; k < 6; ++k) mx = fmaxf(mx, x[k]);
    float s = 0.f;
#pragma unroll
    for (int k = 0; k < 6; ++k) s += __expf(x[k] - mx);
    float lse = mx + __logf(s);
    int lab = cat[idx];
    ce = (lse - x[lab]) * nmask[idx];
  }
  __shared__ float red[256];
  red[threadIdx.x] = ce;
  __syncthreads();
  for (int s2 = 128; s2 > 0; s2 >>= 1) {
    if (threadIdx.x < s2) red[threadIdx.x] += red[threadIdx.x + s2];
    __syncthreads();
  }
  if (threadIdx.x == 0) atomicAdd(accum + 1, red[0]);
}

__global__ void finalize_kernel(const float* __restrict__ accum,
                                float* __restrict__ out) {
  out[0] = sqrtf(accum[0] / (float)NEDGES);
  out[1] = accum[1] / 128.f;
}

extern "C" void kernel_launch(void* const* d_in, const int* in_sizes, int n_in,
                              void* d_out, int out_size, void* d_ws, size_t ws_size,
                              hipStream_t stream) {
  const float* h     = (const float*)d_in[0];
  const float* W1    = (const float*)d_in[1];
  const float* b1    = (const float*)d_in[2];
  const float* W2    = (const float*)d_in[3];
  const float* b2    = (const float*)d_in[4];
  const float* W3    = (const float*)d_in[5];
  const float* b3    = (const float*)d_in[6];
  const float* tgt   = (const float*)d_in[7];
  const float* xhat  = (const float*)d_in[8];
  const float* emask = (const float*)d_in[9];
  const float* nmask = (const float*)d_in[10];
  const int*   cat   = (const int*)d_in[11];

  float* ws    = (float*)d_ws;
  float* Apre  = ws;                       // 8192*64 f32
  float* Bpre  = ws + NPAIRS * DDIM;       // 8192*64 f32
  float* accum = ws + 2 * NPAIRS * DDIM;   // [sq_sum, ce_sum]
  float* out   = (float*)d_out;

  init_kernel<<<1, 1, 0, stream>>>(accum);
  precompute_kernel<<<1024, 32, 0, stream>>>(h, W1, Apre, Bpre);
  edge_kernel<<<1024, 32, 0, stream>>>(Apre, Bpre, b1, W2, b2, W3, b3,
                                       tgt, emask, accum);
  rec_kernel<<<NPAIRS / 256, 256, 0, stream>>>(xhat, cat, nmask, accum);
  finalize_kernel<<<1, 1, 0, stream>>>(accum, out);
}
